// MultiHeadedAttention_30073361007030
// MI455X (gfx1250) — matrix operand order repaired and checked by reading, not compiled, not run on hardware
//
#include <hip/hip_runtime.h>
#include <math.h>

// Problem sizes (fixed by the reference)
#define B_  4
#define S_  1024
#define D_  1024
#define H_  16
#define DH_ 64

typedef __attribute__((ext_vector_type(16))) __bf16          v16bf;
typedef __attribute__((ext_vector_type(8)))  float           v8f;
typedef __attribute__((ext_vector_type(8)))  unsigned short  us8;
typedef __attribute__((ext_vector_type(2)))  __bf16          bf16x2;

union FragB16 {
    unsigned short us[16];
    us8            u8[2];
    bf16x2         h2[8];
    v16bf          v;
};

// Hardware packed f32->bf16 (v_cvt_pk_bf16_f32) when the builtin exists.
__device__ __forceinline__ bf16x2 cvt2(float a, float b) {
#if __has_builtin(__builtin_amdgcn_cvt_pk_bf16_f32)
    return __builtin_amdgcn_cvt_pk_bf16_f32(a, b);
#else
    bf16x2 r;
    r.x = (__bf16)a;
    r.y = (__bf16)b;
    return r;
#endif
}

__device__ __forceinline__ void cvt8(FragB16& f, int j, float4 a, float4 b) {
    f.h2[j + 0] = cvt2(a.x, a.y);
    f.h2[j + 1] = cvt2(a.z, a.w);
    f.h2[j + 2] = cvt2(b.x, b.y);
    f.h2[j + 3] = cvt2(b.z, b.w);
}

__device__ __forceinline__ unsigned short f2bf1(float x) {
    union { bf16x2 v; unsigned short us[2]; } u;
    u.v = cvt2(x, x);
    return u.us[0];
}

// ---------------------------------------------------------------------------
// Kernel 1: projection  Y = scale * (X @ W^T + b), f32 in -> bf16 out.
// One wave per 32(M) x 64(N) tile: 2 A-frags x 4 B-frags -> 8 WMMAs per
// k-step (halves redundant W traffic vs 16-row tiles).
// transpose_out=0 -> (B,H,S,DH); transpose_out=1 -> (B,H,DH,S) (for V).
// ---------------------------------------------------------------------------
__global__ void qkv_proj_kernel(const float* __restrict__ X,
                                const float* __restrict__ W,
                                const float* __restrict__ bias,
                                unsigned short* __restrict__ Y,
                                float scale, int transpose_out) {
    const int lane = threadIdx.x;            // 0..31 (wave32)
    const int m0   = blockIdx.x * 32;        // token-row tile (2 x 16)
    const int n0   = blockIdx.y * 64;        // output-feature tile
    const int klo  = (lane < 16) ? 0 : 8;    // A-frag K base (ISA layout)
    const int kbB  = (lane < 16) ? 0 : 8;    // B-frag K base (ISA layout)

    const float* xrow[2] = {
        X + (size_t)(m0 + (lane & 15)) * D_,
        X + (size_t)(m0 + 16 + (lane & 15)) * D_,
    };

    v8f acc[2][4] = {};
    for (int k0 = 0; k0 < D_; k0 += 32) {
        FragB16 a[2];
#pragma unroll
        for (int mi = 0; mi < 2; ++mi) {
            __builtin_prefetch(xrow[mi] + k0 + 32 + klo, 0, 1);  // speculative: dropped if OOB
            cvt8(a[mi], 0, *(const float4*)(xrow[mi] + k0 + klo),
                           *(const float4*)(xrow[mi] + k0 + klo + 4));
            cvt8(a[mi], 4, *(const float4*)(xrow[mi] + k0 + klo + 16),
                           *(const float4*)(xrow[mi] + k0 + klo + 20));
        }
#pragma unroll
        for (int nt = 0; nt < 4; ++nt) {
            const int ncol = n0 + nt * 16 + (lane & 15);
            const float* wrow = W + (size_t)ncol * D_ + k0 + kbB;
            __builtin_prefetch(wrow + 32, 0, 1);
            FragB16 bf;
            cvt8(bf, 0, *(const float4*)(wrow),     *(const float4*)(wrow + 4));
            cvt8(bf, 4, *(const float4*)(wrow + 16), *(const float4*)(wrow + 20));
#pragma unroll
            for (int mi = 0; mi < 2; ++mi)
                acc[mi][nt] = __builtin_amdgcn_wmma_f32_16x16x32_bf16(
                    false, a[mi].v, false, bf.v, (short)0, acc[mi][nt], false, false);
        }
    }

#pragma unroll
    for (int mi = 0; mi < 2; ++mi) {
#pragma unroll
        for (int nt = 0; nt < 4; ++nt) {
            const int ncol = n0 + nt * 16 + (lane & 15);
            const int h    = ncol >> 6;
            const int dh   = ncol & 63;
            const float bv = bias[ncol];
#pragma unroll
            for (int r = 0; r < 8; ++r) {
                const int mr = m0 + mi * 16 + r + ((lane & 16) ? 8 : 0);  // C layout
                const int bb = mr >> 10;                                   // / S_
                const int ss = mr & (S_ - 1);
                const unsigned short ob = f2bf1((acc[mi][nt][r] + bv) * scale);
                size_t oidx;
                if (transpose_out)
                    oidx = ((size_t)(bb * H_ + h) * DH_ + dh) * (size_t)S_ + ss;
                else
                    oidx = ((size_t)(bb * H_ + h) * S_ + ss) * (size_t)DH_ + dh;
                Y[oidx] = ob;
            }
        }
    }
}

// ---------------------------------------------------------------------------
// Kernel 2: attention for one (b, h, 16-query-row) tile per wave.
//  scores(16x1024) via WMMA -> mask -> LDS -> softmax stats (shfl_xor,
//  wave32) -> attn streamed to global ONCE with non-temporal stores ->
//  context = exp @ V via WMMA with 1/sum folded into the epilogue.
// ---------------------------------------------------------------------------
__global__ void attn_kernel(const unsigned short* __restrict__ Qh,   // (B,H,S,DH) bf16, pre-scaled
                            const unsigned short* __restrict__ Kh,   // (B,H,S,DH) bf16
                            const unsigned short* __restrict__ Vt,   // (B,H,DH,S) bf16
                            const unsigned char*  __restrict__ mask, // (B,S,S) bool
                            float* __restrict__ attn_out,            // (B,H,S,S) f32
                            unsigned short* __restrict__ ctx) {      // (B,S,D) bf16
    extern __shared__ float sc[];            // 16 x S_ scores + 16 row scales
    float* rlds = sc + 16 * S_;

    const int lane = threadIdx.x;
    const int q0   = blockIdx.x * 16;
    const int h    = blockIdx.y;
    const int b    = blockIdx.z;
    const int bh   = b * H_ + h;
    const int klo  = (lane < 16) ? 0 : 8;
    const int kbB  = (lane < 16) ? 0 : 8;

    // Q fragments are invariant over the whole K loop: load once (2 frags).
    const unsigned short* qbase =
        Qh + ((size_t)bh * S_ + (q0 + (lane & 15))) * DH_;
    FragB16 aq[2];
#pragma unroll
    for (int c = 0; c < 2; ++c) {
        aq[c].u8[0] = *(const us8*)(qbase + c * 32 + klo);
        aq[c].u8[1] = *(const us8*)(qbase + c * 32 + klo + 16);
    }

    // ---- Phase 1: scores = Q @ K^T, mask, stash in LDS -------------------
    for (int t = 0; t < S_ / 16; ++t) {
        const int sk0 = t * 16;
        const unsigned short* kb =
            Kh + ((size_t)bh * S_ + sk0 + (lane & 15)) * DH_ + kbB;
        __builtin_prefetch(kb + 16 * DH_, 0, 1);   // next K tile (speculative)
        v8f acc = {};
#pragma unroll
        for (int c = 0; c < 2; ++c) {
            FragB16 bf;
            bf.u8[0] = *(const us8*)(kb + c * 32);
            bf.u8[1] = *(const us8*)(kb + c * 32 + 16);
            acc = __builtin_amdgcn_wmma_f32_16x16x32_bf16(
                false, aq[c].v, false, bf.v, (short)0, acc, false, false);
        }
        const int col = sk0 + (lane & 15);
#pragma unroll
        for (int r = 0; r < 8; ++r) {
            const int row = r + ((lane & 16) ? 8 : 0);
            const size_t midx = ((size_t)b * S_ + (q0 + row)) * (size_t)S_ + col;
            sc[row * S_ + col] = mask[midx] ? -1.0e18f : acc[r];
        }
    }
    __syncthreads();

    // ---- Phase 2: row softmax; stream attn out (non-temporal: written
    //      once, never re-read; keeps the 32MB QKV/ctx set hot in L2) ------
#pragma unroll 1
    for (int r = 0; r < 16; ++r) {
        float m = -3.0e38f;
        for (int c = lane; c < S_; c += 32) m = fmaxf(m, sc[r * S_ + c]);
#pragma unroll
        for (int off = 16; off > 0; off >>= 1) m = fmaxf(m, __shfl_xor(m, off, 32));

        float sum = 0.f;
        for (int c = lane; c < S_; c += 32) {
            float e = __expf(sc[r * S_ + c] - m);
            sc[r * S_ + c] = e;                  // keep unnormalized exp
            sum += e;
        }
#pragma unroll
        for (int off = 16; off > 0; off >>= 1) sum += __shfl_xor(sum, off, 32);
        const float inv = 1.0f / sum;
        if (lane == 0) rlds[r] = inv;

        float* arow = attn_out + ((size_t)bh * S_ + (q0 + r)) * (size_t)S_;
        for (int c = lane; c < S_; c += 32)
            __builtin_nontemporal_store(sc[r * S_ + c] * inv, arow + c);
    }
    __syncthreads();

    // ---- Phase 3: context = exp @ V (V pre-transposed), scale by 1/sum ---
    v8f cacc[4] = {};
    for (int s0 = 0; s0 < S_; s0 += 32) {
        FragB16 a;
        const float* srow = sc + (lane & 15) * S_ + s0 + klo;
        cvt8(a, 0, *(const float4*)(srow),      *(const float4*)(srow + 4));
        cvt8(a, 4, *(const float4*)(srow + 16), *(const float4*)(srow + 20));
#pragma unroll
        for (int nt = 0; nt < 4; ++nt) {
            const int dh = nt * 16 + (lane & 15);
            const unsigned short* vb =
                Vt + ((size_t)bh * DH_ + dh) * (size_t)S_ + s0 + kbB;
            FragB16 bf;
            bf.u8[0] = *(const us8*)(vb);
            bf.u8[1] = *(const us8*)(vb + 16);
            cacc[nt] = __builtin_amdgcn_wmma_f32_16x16x32_bf16(
                false, a.v, false, bf.v, (short)0, cacc[nt], false, false);
        }
    }
#pragma unroll
    for (int nt = 0; nt < 4; ++nt) {
        const int dh = nt * 16 + (lane & 15);
#pragma unroll
        for (int r = 0; r < 8; ++r) {
            const int row = r + ((lane & 16) ? 8 : 0);
            const float val = cacc[nt][r] * rlds[row];
            ctx[((size_t)b * S_ + (q0 + row)) * (size_t)D_ + h * DH_ + dh] = f2bf1(val);
        }
    }
}

// ---------------------------------------------------------------------------
// Kernel 3: output projection  out = ctx(bf16) @ Wo^T + bo  -> f32
// Same 32x64 tiling as the input projections.
// ---------------------------------------------------------------------------
__global__ void out_proj_kernel(const unsigned short* __restrict__ Xb,
                                const float* __restrict__ W,
                                const float* __restrict__ bias,
                                float* __restrict__ Y) {
    const int lane = threadIdx.x;
    const int m0   = blockIdx.x * 32;
    const int n0   = blockIdx.y * 64;
    const int klo  = (lane < 16) ? 0 : 8;
    const int kbB  = (lane < 16) ? 0 : 8;
    const unsigned short* xrow[2] = {
        Xb + (size_t)(m0 + (lane & 15)) * D_,
        Xb + (size_t)(m0 + 16 + (lane & 15)) * D_,
    };

    v8f acc[2][4] = {};
    for (int k0 = 0; k0 < D_; k0 += 32) {
        FragB16 a[2];
#pragma unroll
        for (int mi = 0; mi < 2; ++mi) {
            a[mi].u8[0] = *(const us8*)(xrow[mi] + k0 + klo);
            a[mi].u8[1] = *(const us8*)(xrow[mi] + k0 + klo + 16);
        }
#pragma unroll
        for (int nt = 0; nt < 4; ++nt) {
            const int ncol = n0 + nt * 16 + (lane & 15);
            const float* wrow = W + (size_t)ncol * D_ + k0 + kbB;
            __builtin_prefetch(wrow + 32, 0, 1);
            FragB16 bf;
            cvt8(bf, 0, *(const float4*)(wrow),     *(const float4*)(wrow + 4));
            cvt8(bf, 4, *(const float4*)(wrow + 16), *(const float4*)(wrow + 20));
#pragma unroll
            for (int mi = 0; mi < 2; ++mi)
                acc[mi][nt] = __builtin_amdgcn_wmma_f32_16x16x32_bf16(
                    false, a[mi].v, false, bf.v, (short)0, acc[mi][nt], false, false);
        }
    }
#pragma unroll
    for (int mi = 0; mi < 2; ++mi) {
#pragma unroll
        for (int nt = 0; nt < 4; ++nt) {
            const int ncol = n0 + nt * 16 + (lane & 15);
            const float bb = bias[ncol];
#pragma unroll
            for (int r = 0; r < 8; ++r) {
                const int mr = m0 + mi * 16 + r + ((lane & 16) ? 8 : 0);
                Y[(size_t)mr * D_ + ncol] = acc[mi][nt][r] + bb;
            }
        }
    }
}

// ---------------------------------------------------------------------------
extern "C" void kernel_launch(void* const* d_in, const int* in_sizes, int n_in,
                              void* d_out, int out_size, void* d_ws, size_t ws_size,
                              hipStream_t stream) {
    const float*         key   = (const float*)d_in[0];
    const float*         value = (const float*)d_in[1];
    const float*         query = (const float*)d_in[2];
    const unsigned char* mask  = (const unsigned char*)d_in[3];  // bool array
    const float* Wk = (const float*)d_in[4];
    const float* bk = (const float*)d_in[5];
    const float* Wv = (const float*)d_in[6];
    const float* bv = (const float*)d_in[7];
    const float* Wq = (const float*)d_in[8];
    const float* bq = (const float*)d_in[9];
    const float* Wo = (const float*)d_in[10];
    const float* bo = (const float*)d_in[11];

    float* out_f    = (float*)d_out;                       // (B,S,D)
    float* attn_out = out_f + (size_t)B_ * S_ * D_;        // (B,H,S,S)

    // Workspace: 4 bf16 tensors of B*S*D each (8 MB apiece, 32 MB total)
    unsigned short* Qh  = (unsigned short*)d_ws;
    unsigned short* Kh  = Qh + (size_t)B_ * S_ * D_;
    unsigned short* Vt  = Kh + (size_t)B_ * S_ * D_;
    unsigned short* ctx = Vt + (size_t)B_ * S_ * D_;

    dim3 pgrid(B_ * S_ / 32, D_ / 64);
    qkv_proj_kernel<<<pgrid, 32, 0, stream>>>(query, Wq, bq, Qh, 0.125f, 0);
    qkv_proj_kernel<<<pgrid, 32, 0, stream>>>(key,   Wk, bk, Kh, 1.0f,   0);
    qkv_proj_kernel<<<pgrid, 32, 0, stream>>>(value, Wv, bv, Vt, 1.0f,   1);

    dim3 agrid(S_ / 16, H_, B_);
    size_t shbytes = (size_t)16 * S_ * sizeof(float) + 16 * sizeof(float);
    attn_kernel<<<agrid, 32, shbytes, stream>>>(Qh, Kh, Vt, mask, attn_out, ctx);

    out_proj_kernel<<<pgrid, 32, 0, stream>>>(ctx, Wo, bo, out_f);
}
